// MPNNLayer_59184649339500
// MI455X (gfx1250) — compile-verified, matrix operand order checked by reading, not run on hardware
//
#include <hip/hip_runtime.h>
#include <hip/hip_bf16.h>
#include <stdint.h>

// ---------------------------------------------------------------------------
// MPNN layer on gfx1250: bf16 WMMA (v_wmma_f32_16x16x32_bf16) for both MLPs.
// Roofline: ~170 GFLOP vs ~1.08 GB HBM traffic -> matrix pipes + wide staging.
// ---------------------------------------------------------------------------

typedef __attribute__((ext_vector_type(16))) __bf16 v16bf;
typedef __attribute__((ext_vector_type(8)))  float  v8f;

struct Frag32 { uint4 lo; uint4 hi; };   // 32B = one wmma A/B operand per lane

__device__ __forceinline__ unsigned short f2bf(float f) {
  unsigned u = __float_as_uint(f);
  u += 0x7FFFu + ((u >> 16) & 1u);       // round-to-nearest-even
  return (unsigned short)(u >> 16);
}
__device__ __forceinline__ unsigned pack2(float a, float b) {
  return (unsigned)f2bf(a) | ((unsigned)f2bf(b) << 16);
}
__device__ __forceinline__ uint2 pack4(float4 v) {
  uint2 p; p.x = pack2(v.x, v.y); p.y = pack2(v.z, v.w); return p;
}

// ---------------------------------------------------------------------------
// Workspace layout (bytes)
// ---------------------------------------------------------------------------
#define WS_W1F   0          // eW1 frags: 9kt*8nt*32lane*8dw = 18432 dw = 73728 B
#define WS_W2F   73728      // eW2 frags: 4*8*32*8 = 8192 dw = 32768 B
#define WS_NW1F  106496     // nW1 frags: 8*8*32*8 = 16384 dw = 65536 B
#define WS_NW2F  172032     // nW2 frags: 8192 dw = 32768 B
#define WS_BIAS  204800     // eb1|eb2|nb1|nb2 : 512 f32 = 2048 B
#define WS_MI    206848     // m_i : B*128 f32

// ---------------------------------------------------------------------------
// Prep: swizzle f32 weights into WMMA B-fragment-major bf16-pair layout.
// dst dword index i = kt*2048 + nt*256 + lane*8 + j holds packed
// (W[kk][n], W[kk+1][n]) with kk = kt*32 + (j&4?16:0) + (lane>>4)*8 + (j&3)*2,
// n = nt*16 + (lane&15)   (matches ISA 16-bit B operand layout, wave32).
// ---------------------------------------------------------------------------
__device__ void swizzleW(const float* __restrict__ W, int Kt,
                         unsigned* __restrict__ dst, int g, int stride) {
  int total = Kt * 2048;
  for (int i = g; i < total; i += stride) {
    int kt   = i >> 11;
    int nt   = (i >> 8) & 7;
    int lane = (i >> 3) & 31;
    int j    = i & 7;
    int kk = kt * 32 + ((j & 4) ? 16 : 0) + ((lane >> 4) * 8) + ((j & 3) * 2);
    int n  = nt * 16 + (lane & 15);
    dst[i] = pack2(W[kk * 128 + n], W[(kk + 1) * 128 + n]);
  }
}

__global__ void prep_kernel(const float* __restrict__ eW1, const float* __restrict__ eW2,
                            const float* __restrict__ nW1, const float* __restrict__ nW2,
                            const float* __restrict__ eb1, const float* __restrict__ eb2,
                            const float* __restrict__ nb1, const float* __restrict__ nb2,
                            unsigned* __restrict__ w1f, unsigned* __restrict__ w2f,
                            unsigned* __restrict__ nw1f, unsigned* __restrict__ nw2f,
                            float* __restrict__ bias) {
  int g = blockIdx.x * blockDim.x + threadIdx.x;
  int stride = gridDim.x * blockDim.x;
  swizzleW(eW1, 9, w1f, g, stride);   // 288x128
  swizzleW(eW2, 4, w2f, g, stride);   // 128x128
  swizzleW(nW1, 8, nw1f, g, stride);  // 256x128
  swizzleW(nW2, 4, nw2f, g, stride);  // 128x128
  for (int i = g; i < 128; i += stride) {
    bias[i]       = eb1[i];
    bias[128 + i] = eb2[i];
    bias[256 + i] = nb1[i];
    bias[384 + i] = nb2[i];
  }
}

// ---------------------------------------------------------------------------
// Edge kernel LDS layout (dynamic, 185856 B)
// ---------------------------------------------------------------------------
#define E_OW1   0          // 73728
#define E_OW2   73728      // 32768
#define E_OB    106496     // 1024 (eb1,eb2)
#define E_OMSK  107520     // 512  (128 mask f32)
#define E_OMAC  108032     // 2048 (4*128 f32 m_acc)
#define E_OA    110080     // 8 waves * 16 rows * 592 B = 75776
#define E_AROW  592        // 288 bf16 + pad, 16B aligned
#define E_AWAVE (16 * 592)
#define E_H1ROW 272        // 128 bf16 + pad
#define EDGE_SMEM (E_OA + 8 * E_AWAVE)   // 185856

__device__ __forceinline__ v16bf load_frag2(const char* p0, const char* p1) {
  Frag32 f;
  f.lo = *(const uint4*)p0;
  f.hi = *(const uint4*)p1;
  return __builtin_bit_cast(v16bf, f);
}

__launch_bounds__(256, 1)
__global__ void edge_kernel(const float* __restrict__ h_self,
                            const float* __restrict__ h_nei,
                            const float* __restrict__ e_ij,
                            const float* __restrict__ mask,
                            const unsigned* __restrict__ w1f,
                            const unsigned* __restrict__ w2f,
                            const float* __restrict__ bias,   // eb1(128)|eb2(128)
                            float* __restrict__ mi) {
  extern __shared__ char smem[];
  unsigned* s_w1  = (unsigned*)(smem + E_OW1);
  unsigned* s_w2  = (unsigned*)(smem + E_OW2);
  float*    s_b   = (float*)(smem + E_OB);
  float*    s_msk = (float*)(smem + E_OMSK);
  float*    s_mac = (float*)(smem + E_OMAC);

  const int tid  = threadIdx.x;
  const int b0   = blockIdx.x * 4;          // 4 batch rows per block
  const int row0 = blockIdx.x * 128;        // 128 (b,k) rows per block

  // --- stage weights/bias/mask, zero m_acc -------------------------------
  {
    const uint4* s1 = (const uint4*)w1f;  uint4* d1 = (uint4*)s_w1;
    for (int i = tid; i < 18432 / 4; i += 256) d1[i] = s1[i];
    const uint4* s2 = (const uint4*)w2f;  uint4* d2 = (uint4*)s_w2;
    for (int i = tid; i < 8192 / 4; i += 256) d2[i] = s2[i];
    s_b[tid] = bias[tid];                    // 256 floats
    if (tid < 128) s_msk[tid] = mask[row0 + tid];
    s_mac[tid] = 0.f; s_mac[tid + 256] = 0.f;
  }
  // --- stage A rows as bf16: [h_self | h_nei | e_ij] ---------------------
  // float4 global loads -> packed bf16 pairs -> ds_store_b64
  for (int i = tid; i < 128 * 32; i += 256) {      // h_self (broadcast per b)
    int r = i >> 5, g4 = i & 31;
    float4 v = *(const float4*)(h_self + (b0 + (r >> 5)) * 128 + g4 * 4);
    *(uint2*)(smem + E_OA + r * E_AROW + g4 * 8) = pack4(v);
  }
  for (int i = tid; i < 128 * 32; i += 256) {      // h_nei
    int r = i >> 5, g4 = i & 31;
    float4 v = *(const float4*)(h_nei + (size_t)(row0 + r) * 128 + g4 * 4);
    *(uint2*)(smem + E_OA + r * E_AROW + 256 + g4 * 8) = pack4(v);
  }
  for (int i = tid; i < 128 * 8; i += 256) {       // e_ij
    int r = i >> 3, g4 = i & 7;
    float4 v = *(const float4*)(e_ij + (size_t)(row0 + r) * 32 + g4 * 4);
    *(uint2*)(smem + E_OA + r * E_AROW + 512 + g4 * 8) = pack4(v);
  }
  __syncthreads();

  const int w    = tid >> 5;
  const int lane = tid & 31;
  const int hf   = lane >> 4;     // half-wave (selects K/M sub-block)
  const int rm   = lane & 15;
  const char* Abase = smem + E_OA + w * E_AWAVE;

  // --- GEMM1: [16x288] @ eW1 -> [16x128] ---------------------------------
  v8f acc[8];
  #pragma unroll
  for (int nt = 0; nt < 8; ++nt) acc[nt] = (v8f){0,0,0,0,0,0,0,0};

  #pragma unroll
  for (int kt = 0; kt < 9; ++kt) {
    const char* ap = Abase + rm * E_AROW + kt * 64 + hf * 16;
    v16bf A = load_frag2(ap, ap + 32);
    #pragma unroll
    for (int nt = 0; nt < 8; ++nt) {
      const char* bp = (const char*)s_w1 + (((kt * 8 + nt) * 32 + lane) * 32);
      v16bf Bf = load_frag2(bp, bp + 16);
      acc[nt] = __builtin_amdgcn_wmma_f32_16x16x32_bf16(
          false, A, false, Bf, (short)0, acc[nt], false, false);
    }
  }
  // bias + ReLU -> h1 staged bf16 (reuse this wave's A region)
  #pragma unroll
  for (int nt = 0; nt < 8; ++nt) {
    float bs = s_b[nt * 16 + rm];
    #pragma unroll
    for (int r = 0; r < 8; ++r) {
      float h = acc[nt][r] + bs;
      h = h > 0.f ? h : 0.f;
      int rr = hf * 8 + r;                         // C layout: M = r + 8*hf
      *(unsigned short*)(Abase + rr * E_H1ROW + (nt * 16 + rm) * 2) = f2bf(h);
    }
  }
  __syncthreads();

  // --- GEMM2: [16x128] @ eW2 -> [16x128] ---------------------------------
  v8f acc2[8];
  #pragma unroll
  for (int nt = 0; nt < 8; ++nt) acc2[nt] = (v8f){0,0,0,0,0,0,0,0};
  #pragma unroll
  for (int kt = 0; kt < 4; ++kt) {
    const char* ap = Abase + rm * E_H1ROW + kt * 64 + hf * 16;
    v16bf A = load_frag2(ap, ap + 32);
    #pragma unroll
    for (int nt = 0; nt < 8; ++nt) {
      const char* bp = (const char*)s_w2 + (((kt * 8 + nt) * 32 + lane) * 32);
      v16bf Bf = load_frag2(bp, bp + 16);
      acc2[nt] = __builtin_amdgcn_wmma_f32_16x16x32_bf16(
          false, A, false, Bf, (short)0, acc2[nt], false, false);
    }
  }
  // bias + ReLU, multiply by mask, reduce over K into LDS m_acc
  float mreg[8];
  #pragma unroll
  for (int r = 0; r < 8; ++r) mreg[r] = s_msk[w * 16 + hf * 8 + r];
  const int bloc = w >> 1;                         // 32 k-rows per b = 2 waves
  #pragma unroll
  for (int nt = 0; nt < 8; ++nt) {
    float bs = s_b[128 + nt * 16 + rm];
    float p = 0.f;
    #pragma unroll
    for (int r = 0; r < 8; ++r) {
      float h = acc2[nt][r] + bs;
      h = h > 0.f ? h : 0.f;
      p += h * mreg[r];
    }
    atomicAdd(&s_mac[bloc * 128 + nt * 16 + rm], p);  // ds_add_f32
  }
  __syncthreads();
  for (int i = tid; i < 512; i += 256)
    mi[(size_t)blockIdx.x * 512 + i] = s_mac[i];
}

// ---------------------------------------------------------------------------
// Node kernel LDS layout (dynamic, 166912 B)
// ---------------------------------------------------------------------------
#define N_OW1   0          // 65536
#define N_OW2   65536      // 32768
#define N_OB    98304      // 1024 (nb1,nb2)
#define N_OA    99328      // 8 * 16 * 528 = 67584
#define N_AROW  528        // 256 bf16 + pad
#define N_AWAVE (16 * 528)
#define N_H1ROW 272
#define NODE_SMEM (N_OA + 8 * N_AWAVE)   // 166912

__launch_bounds__(256, 1)
__global__ void node_kernel(const float* __restrict__ h_self,
                            const float* __restrict__ mi,
                            const unsigned* __restrict__ nw1f,
                            const unsigned* __restrict__ nw2f,
                            const float* __restrict__ bias,   // nb1(128)|nb2(128)
                            float* __restrict__ out, int B) {
  extern __shared__ char smem[];
  unsigned* s_w1 = (unsigned*)(smem + N_OW1);
  unsigned* s_w2 = (unsigned*)(smem + N_OW2);
  float*    s_b  = (float*)(smem + N_OB);

  const int tid  = threadIdx.x;
  const int row0 = blockIdx.x * 128;

  {
    const uint4* s1 = (const uint4*)nw1f; uint4* d1 = (uint4*)s_w1;
    for (int i = tid; i < 16384 / 4; i += 256) d1[i] = s1[i];
    const uint4* s2 = (const uint4*)nw2f; uint4* d2 = (uint4*)s_w2;
    for (int i = tid; i < 8192 / 4; i += 256) d2[i] = s2[i];
    s_b[tid] = bias[tid];
  }
  // stage A rows: [h_self | m_i] (clamp tail rows; stores are guarded later)
  for (int i = tid; i < 128 * 32; i += 256) {
    int r = i >> 5, g4 = i & 31;
    int rr = row0 + r; if (rr >= B) rr = B - 1;
    float4 v = *(const float4*)(h_self + (size_t)rr * 128 + g4 * 4);
    *(uint2*)(smem + N_OA + r * N_AROW + g4 * 8) = pack4(v);
  }
  for (int i = tid; i < 128 * 32; i += 256) {
    int r = i >> 5, g4 = i & 31;
    int rr = row0 + r; if (rr >= B) rr = B - 1;
    float4 v = *(const float4*)(mi + (size_t)rr * 128 + g4 * 4);
    *(uint2*)(smem + N_OA + r * N_AROW + 256 + g4 * 8) = pack4(v);
  }
  __syncthreads();

  const int w    = tid >> 5;
  const int lane = tid & 31;
  const int hf   = lane >> 4;
  const int rm   = lane & 15;
  const char* Abase = smem + N_OA + w * N_AWAVE;

  v8f acc[8];
  #pragma unroll
  for (int nt = 0; nt < 8; ++nt) acc[nt] = (v8f){0,0,0,0,0,0,0,0};
  #pragma unroll
  for (int kt = 0; kt < 8; ++kt) {                 // K = 256
    const char* ap = Abase + rm * N_AROW + kt * 64 + hf * 16;
    v16bf A = load_frag2(ap, ap + 32);
    #pragma unroll
    for (int nt = 0; nt < 8; ++nt) {
      const char* bp = (const char*)s_w1 + (((kt * 8 + nt) * 32 + lane) * 32);
      v16bf Bf = load_frag2(bp, bp + 16);
      acc[nt] = __builtin_amdgcn_wmma_f32_16x16x32_bf16(
          false, A, false, Bf, (short)0, acc[nt], false, false);
    }
  }
  #pragma unroll
  for (int nt = 0; nt < 8; ++nt) {
    float bs = s_b[nt * 16 + rm];
    #pragma unroll
    for (int r = 0; r < 8; ++r) {
      float h = acc[nt][r] + bs;
      h = h > 0.f ? h : 0.f;
      int rr = hf * 8 + r;
      *(unsigned short*)(Abase + rr * N_H1ROW + (nt * 16 + rm) * 2) = f2bf(h);
    }
  }
  __syncthreads();

  v8f acc2[8];
  #pragma unroll
  for (int nt = 0; nt < 8; ++nt) acc2[nt] = (v8f){0,0,0,0,0,0,0,0};
  #pragma unroll
  for (int kt = 0; kt < 4; ++kt) {                 // K = 128
    const char* ap = Abase + rm * N_H1ROW + kt * 64 + hf * 16;
    v16bf A = load_frag2(ap, ap + 32);
    #pragma unroll
    for (int nt = 0; nt < 8; ++nt) {
      const char* bp = (const char*)s_w2 + (((kt * 8 + nt) * 32 + lane) * 32);
      v16bf Bf = load_frag2(bp, bp + 16);
      acc2[nt] = __builtin_amdgcn_wmma_f32_16x16x32_bf16(
          false, A, false, Bf, (short)0, acc2[nt], false, false);
    }
  }
  #pragma unroll
  for (int nt = 0; nt < 8; ++nt) {
    float bs = s_b[128 + nt * 16 + rm];
    #pragma unroll
    for (int r = 0; r < 8; ++r) {
      float h = acc2[nt][r] + bs;
      h = h > 0.f ? h : 0.f;
      int row = row0 + w * 16 + hf * 8 + r;
      if (row < B) out[(size_t)row * 128 + nt * 16 + rm] = h;
    }
  }
}

// ---------------------------------------------------------------------------
extern "C" void kernel_launch(void* const* d_in, const int* in_sizes, int n_in,
                              void* d_out, int out_size, void* d_ws, size_t ws_size,
                              hipStream_t stream) {
  (void)n_in; (void)out_size; (void)ws_size;
  const float* h_self = (const float*)d_in[0];
  const float* h_nei  = (const float*)d_in[1];
  const float* e_ij   = (const float*)d_in[2];
  const float* mask   = (const float*)d_in[3];
  const float* eW1 = (const float*)d_in[4];
  const float* eb1 = (const float*)d_in[5];
  const float* eW2 = (const float*)d_in[6];
  const float* eb2 = (const float*)d_in[7];
  const float* nW1 = (const float*)d_in[8];
  const float* nb1 = (const float*)d_in[9];
  const float* nW2 = (const float*)d_in[10];
  const float* nb2 = (const float*)d_in[11];
  float* out = (float*)d_out;

  const int B = in_sizes[0] / 128;   // 50000 (divisible by 4)

  char* ws = (char*)d_ws;
  unsigned* w1f  = (unsigned*)(ws + WS_W1F);
  unsigned* w2f  = (unsigned*)(ws + WS_W2F);
  unsigned* nw1f = (unsigned*)(ws + WS_NW1F);
  unsigned* nw2f = (unsigned*)(ws + WS_NW2F);
  float*    bias = (float*)(ws + WS_BIAS);
  float*    mi   = (float*)(ws + WS_MI);

  hipFuncSetAttribute((const void*)edge_kernel,
                      hipFuncAttributeMaxDynamicSharedMemorySize, EDGE_SMEM);
  hipFuncSetAttribute((const void*)node_kernel,
                      hipFuncAttributeMaxDynamicSharedMemorySize, NODE_SMEM);

  prep_kernel<<<64, 256, 0, stream>>>(eW1, eW2, nW1, nW2, eb1, eb2, nb1, nb2,
                                      w1f, w2f, nw1f, nw2f, bias);
  edge_kernel<<<B / 4, 256, EDGE_SMEM, stream>>>(h_self, h_nei, e_ij, mask,
                                                 w1f, w2f, bias, mi);
  node_kernel<<<(B + 127) / 128, 256, NODE_SMEM, stream>>>(h_self, mi, nw1f,
                                                           nw2f, bias + 256,
                                                           out, B);
}
